// Encoder_42494406427277
// MI455X (gfx1250) — matrix operand order repaired and checked by reading, not compiled, not run on hardware
//
#include <hip/hip_runtime.h>

typedef __attribute__((ext_vector_type(16))) _Float16 v16h;
typedef __attribute__((ext_vector_type(8)))  _Float16 v8h;
typedef __attribute__((ext_vector_type(8)))  float    v8f;

#define S_TOK 6561
#define S_PAD 6576
#define DMODEL 34
#define HW 81

// ---------------- WMMA helpers (gfx1250 wave32, 16x16x32 f16) ----------------

__device__ inline v8f wmma_f16(v16h a, v16h b, v8f c) {
  return __builtin_amdgcn_wmma_f32_16x16x32_f16(
      false, a, false, b, (short)0, c, false, false);
}

// A fragment from a row-major f16 row pointer already offset by m*stride + 16*g.
// Layout: lane(g,m) holds k = {16g..16g+15}.
__device__ inline v16h load_a16(const _Float16* p) {
  v8h lo = *(const v8h*)p;
  v8h hi = *(const v8h*)(p + 8);
  return __builtin_shufflevector(lo, hi, 0, 1, 2, 3, 4, 5, 6, 7, 8, 9, 10, 11,
                                 12, 13, 14, 15);
}

// A fragment from an LDS tile stored row-major [16][stride].
__device__ inline v16h pack_a(const _Float16* lds, int stride, int lane) {
  int g = (lane >> 4) & 1, m = lane & 15;
  return load_a16(lds + m * stride + 16 * g);
}

// B fragment: 32x16 (KxN) from LDS tile stored as [n][32k].
__device__ inline v16h pack_b(const _Float16* lds, int lane) {
  int g = (lane >> 4) & 1, n = lane & 15;
  const _Float16* row = lds + n * 32 + 16 * g;
  v16h b;
#pragma unroll
  for (int i = 0; i < 16; ++i) b[i] = row[i];
  return b;
}

// B fragment from a V tile stored as [key][d] (16x32); K-dim = key (pad 16->32).
__device__ inline v16h pack_bv(const _Float16* ldsV, int lane, int dbase) {
  int g = (lane >> 4) & 1, n = lane & 15;
  v16h b;
#pragma unroll
  for (int i = 0; i < 16; ++i)
    b[i] = g ? (_Float16)0.f : ldsV[i * 32 + dbase + n];
  return b;
}

// Async copy: 32 lanes x 16B contiguous chunk, global -> LDS (ASYNCcnt).
__device__ inline void async_b128(unsigned ldsByteOff, const _Float16* g) {
  asm volatile("global_load_async_to_lds_b128 %0, %1, off" ::"v"(ldsByteOff),
               "v"(g)
               : "memory");
}
__device__ inline void wait_async_le14() {
  asm volatile("s_wait_asynccnt 14" ::: "memory");
}

// ---------------- conv1 + conv2 + argmax ----------------

__global__ __launch_bounds__(256) void conv_kernel(
    const float* __restrict__ obs, const float* __restrict__ w1,
    const float* __restrict__ b1, const float* __restrict__ w2,
    const float* __restrict__ b2, float* __restrict__ cell,
    int* __restrict__ agent) {
  __shared__ float mid[16][HW];
  int tid = threadIdx.x;
  for (int idx = tid; idx < 16 * HW; idx += 256) {
    int c = idx / HW, p = idx % HW, r = p / 9, q = p % 9;
    float s = b1[c];
    for (int ic = 0; ic < 2; ++ic)
      for (int kr = 0; kr < 3; ++kr)
        for (int kc = 0; kc < 3; ++kc) {
          int rr = r + kr - 1, cc = q + kc - 1;
          if (rr >= 0 && rr < 9 && cc >= 0 && cc < 9)
            s += obs[ic * HW + rr * 9 + cc] * w1[((c * 2 + ic) * 3 + kr) * 3 + kc];
        }
    mid[c][p] = fmaxf(s, 0.f);
  }
  __syncthreads();
  for (int idx = tid; idx < 16 * HW; idx += 256) {
    int c = idx / HW, p = idx % HW, r = p / 9, q = p % 9;
    float s = b2[c];
    for (int ic = 0; ic < 16; ++ic)
      for (int kr = 0; kr < 3; ++kr)
        for (int kc = 0; kc < 3; ++kc) {
          int rr = r + kr - 1, cc = q + kc - 1;
          if (rr >= 0 && rr < 9 && cc >= 0 && cc < 9)
            s += mid[ic][rr * 9 + cc] * w2[((c * 16 + ic) * 3 + kr) * 3 + kc];
        }
    cell[p * 16 + c] = fmaxf(s, 0.f);
  }
  if (tid == 0) {
    int best = 0;
    float bv = obs[0];
    for (int i = 1; i < HW; ++i) {
      float v = obs[i];
      if (v > bv) { bv = v; best = i; }
    }
    *agent = best;
  }
}

// ---------------- weight pre-swizzle: W[N,K] f32 -> B-fragment f16 blocks -----
// out[((t*Ksteps+s)*32 + lane)*16 + i] = W[t*16 + (lane&15)][s*32 + 16*(lane>>4) + i]

__global__ void prep_w_kernel(const float* __restrict__ W,
                              _Float16* __restrict__ out, int N, int K,
                              int total) {
  int idx = blockIdx.x * 256 + threadIdx.x;
  if (idx >= total) return;
  int i = idx & 15;
  int lane = (idx >> 4) & 31;
  int blk = idx >> 9;  // t*Ksteps + s, Ksteps baked into addressing by caller
  // caller passes total = tilesN*Ksteps*512; recover s,t via K:
  int Ksteps = (K + 31) / 32;
  int s = blk % Ksteps, t = blk / Ksteps;
  int n = t * 16 + (lane & 15);
  int k = s * 32 + 16 * (lane >> 4) + i;
  out[idx] = (_Float16)((n < N && k < K) ? W[(size_t)n * K + k] : 0.f);
}

// ---------------- token build: X f32 [S,34] + Xh f16 padded [S_PAD,64] -------

__global__ void tokens_kernel(const float* __restrict__ cell,
                              float* __restrict__ X,
                              _Float16* __restrict__ Xh) {
  int idx = blockIdx.x * blockDim.x + threadIdx.x;
  if (idx >= S_PAD * 64) return;
  int t = idx >> 6, d = idx & 63;
  float v = 0.f;
  if (t < S_TOK && d < DMODEL) {
    int i = t / HW, j = t % HW;
    if (d < 16)       v = cell[i * 16 + d];
    else if (d < 32)  v = cell[j * 16 + (d - 16)];
    else if (d == 32) v = ((float)(i / 9) - (float)(j / 9)) * 0.25f;
    else              v = ((float)(i % 9) - (float)(j % 9)) * 0.25f;
    X[t * DMODEL + d] = v;
  }
  Xh[idx] = (_Float16)v;
}

__global__ void zeroh_kernel(_Float16* __restrict__ p, int n) {
  int i = blockIdx.x * 256 + threadIdx.x;
  if (i < n) p[i] = (_Float16)0.f;
}

// ---------------- GEMM: C[M,N] = Ah[Mpad,64-padded] @ Ws^T + bias ------------
// Ah: padded f16 activations. Ws: pre-swizzled B blocks. No LDS, no barriers.

__global__ __launch_bounds__(256) void gemm_kernel(
    const _Float16* __restrict__ Ah, const _Float16* __restrict__ Ws,
    const float* __restrict__ bias, float* __restrict__ C, int M, int N,
    int Ksteps, int relu) {
  int lane = threadIdx.x & 31;
  int wave = threadIdx.x >> 5;
  int tilesN = (N + 15) >> 4;
  int tilesM = (M + 15) >> 4;
  int tile = blockIdx.x * 8 + wave;
  if (tile >= tilesM * tilesN) return;
  int tm = tile / tilesN, tn = tile % tilesN;
  int g = lane >> 4, n = lane & 15;

  const _Float16* Arow = Ah + (size_t)(tm * 16 + n) * 64 + 16 * g;
  const _Float16* Bp = Ws + ((size_t)tn * Ksteps * 32 + lane) * 16;

  v8f acc = {};
  for (int s = 0; s < Ksteps; ++s) {
    if (s + 1 < Ksteps) __builtin_prefetch(Bp + 512, 0, 0);
    v16h av = load_a16(Arow + s * 32);
    v16h bv = *(const v16h*)Bp;
    acc = wmma_f16(av, bv, acc);
    Bp += 512;
  }
  int col = tn * 16 + n;
  float bb = (col < N) ? bias[col] : 0.f;
#pragma unroll
  for (int v = 0; v < 8; ++v) {
    int row = tm * 16 + v + 8 * g;
    if (row < M && col < N) {
      float r = acc[v] + bb;
      if (relu) r = fmaxf(r, 0.f);
      C[(size_t)row * N + col] = r;
    }
  }
}

// ---------------- flash attention: one wave per (head, 16-query tile) --------
// qkv f32 [S,102]; output: padded f16 [S_PAD,64] (pre-zeroed buffer).

__global__ __launch_bounds__(32) void attn_kernel(const float* __restrict__ qkv,
                                                  _Float16* __restrict__ oh,
                                                  int S) {
  __shared__ _Float16 ldsQ[16 * 32];
  __shared__ _Float16 ldsK[16 * 32];
  __shared__ _Float16 ldsV[16 * 32];
  __shared__ _Float16 ldsP[16 * 32];
  __shared__ float ldsS[16 * 16];
  __shared__ float sM[16], sL[16], sAl[16];

  int lane = threadIdx.x;
  int qt = blockIdx.x >> 1;
  int h = blockIdx.x & 1;
  const float scale = 0.2425356250363330f;  // 1/sqrt(17)

  for (int r = 0; r < 16; ++r) {
    int q = qt * 16 + r;
    float v = (q < S && lane < 17) ? qkv[(size_t)q * 102 + h * 17 + lane] * scale
                                   : 0.f;
    ldsQ[r * 32 + lane] = (_Float16)v;
    ldsP[r * 32 + lane] = (_Float16)0.f;
  }
  if (lane < 16) { sM[lane] = -1e30f; sL[lane] = 0.f; }
  __syncthreads();

  v16h qa = pack_a(ldsQ, 32, lane);
  v8f accLo = {}, accHi = {};

  int nkt = (S + 15) >> 4;
  for (int kt = 0; kt < nkt; ++kt) {
    for (int r = 0; r < 16; ++r) {
      int key = kt * 16 + r;
      bool ok = (key < S) && (lane < 17);
      ldsK[r * 32 + lane] =
          (_Float16)(ok ? qkv[(size_t)key * 102 + 34 + h * 17 + lane] : 0.f);
      ldsV[r * 32 + lane] =
          (_Float16)(ok ? qkv[(size_t)key * 102 + 68 + h * 17 + lane] : 0.f);
    }
    __syncthreads();

    v16h kb = pack_b(ldsK, lane);
    v8f sc = {};
    sc = wmma_f16(qa, kb, sc);
    {
      int g = lane >> 4, n = lane & 15;
#pragma unroll
      for (int v = 0; v < 8; ++v) ldsS[(v + 8 * g) * 16 + n] = sc[v];
    }
    __syncthreads();

    if (lane < 16) {
      int r = lane;
      int valid = S - kt * 16;
      if (valid > 16) valid = 16;
      float mOld = sM[r], mNew = mOld;
      for (int j = 0; j < valid; ++j) mNew = fmaxf(mNew, ldsS[r * 16 + j]);
      float alpha = __expf(mOld - mNew);
      float psum = 0.f;
      for (int j = 0; j < 16; ++j) {
        float p = (j < valid) ? __expf(ldsS[r * 16 + j] - mNew) : 0.f;
        ldsP[r * 32 + j] = (_Float16)p;
        psum += p;
      }
      sL[r] = sL[r] * alpha + psum;
      sM[r] = mNew;
      sAl[r] = alpha;
    }
    __syncthreads();

    {
      int g = lane >> 4;
#pragma unroll
      for (int v = 0; v < 8; ++v) {
        float a_ = sAl[v + 8 * g];
        accLo[v] *= a_;
        accHi[v] *= a_;
      }
    }
    v16h pa = pack_a(ldsP, 32, lane);
    v16h vbLo = pack_bv(ldsV, lane, 0);
    v16h vbHi = pack_bv(ldsV, lane, 16);
    accLo = wmma_f16(pa, vbLo, accLo);
    accHi = wmma_f16(pa, vbHi, accHi);
    __syncthreads();
  }

  int g = lane >> 4, n = lane & 15;
#pragma unroll
  for (int v = 0; v < 8; ++v) {
    int rl = v + 8 * g;
    int row = qt * 16 + rl;
    if (row < S) {
      float inv = 1.f / sL[rl];
      oh[(size_t)row * 64 + h * 17 + n] = (_Float16)(accLo[v] * inv);
      if (n == 0)
        oh[(size_t)row * 64 + h * 17 + 16] = (_Float16)(accHi[v] * inv);
    }
  }
}

// ---------------- fused FFN with async double-buffered weight tiles ----------
// out = relu(X1h @ W1^T + b1) @ W2^T + b2.  W1s: [128][2][512], W2s: [3][64][512].

__global__ __launch_bounds__(32) void ffn_kernel(
    const _Float16* __restrict__ Xh, const _Float16* __restrict__ W1s,
    const float* __restrict__ b1, const _Float16* __restrict__ W2s,
    const float* __restrict__ b2, float* __restrict__ out, int S) {
  __shared__ __align__(32) _Float16 ldsW[2][7 * 512];  // 7 x 1KB blocks / buf
  __shared__ __align__(32) _Float16 ldsMid[16 * 32];
  int lane = threadIdx.x;
  int tm = blockIdx.x;
  int g = lane >> 4, nn = lane & 15;

  const _Float16* Arow = Xh + (size_t)(tm * 16 + nn) * 64 + 16 * g;
  v16h xa0 = load_a16(Arow);
  v16h xa1 = load_a16(Arow + 32);

  unsigned ldsBase = (unsigned)(size_t)(&ldsW[0][0]) + lane * 16;

  v8f acc2[3];
  acc2[0] = (v8f){};
  acc2[1] = (v8f){};
  acc2[2] = (v8f){};

  // issue 14 async 512B chunks for iteration `it` into buffer `buf`
  auto issue_tiles = [&](int it, int buf) {
    unsigned b = ldsBase + buf * (7 * 1024);
    const _Float16* w1p = W1s + (size_t)it * 2048 + lane * 8;  // 4 blocks, 4KB
#pragma unroll
    for (int p = 0; p < 8; ++p) async_b128(b + p * 512, w1p + p * 256);
#pragma unroll
    for (int nt2 = 0; nt2 < 3; ++nt2) {
      const _Float16* w2p = W2s + (size_t)(nt2 * 64 + it) * 512 + lane * 8;
      async_b128(b + (8 + 2 * nt2) * 512, w2p);
      async_b128(b + (9 + 2 * nt2) * 512, w2p + 256);
    }
  };

  issue_tiles(0, 0);
  for (int it = 0; it < 64; ++it) {
    int cur = it & 1;
    issue_tiles((it + 1) & 63, cur ^ 1);  // prefetch next (wraps harmlessly)
    wait_async_le14();                    // current buffer's 14 chunks done

    const _Float16* wb = &ldsW[cur][0] + lane * 16;
#pragma unroll
    for (int half = 0; half < 2; ++half) {
      int nt = it * 2 + half;
      v8f acc = {};
      acc = wmma_f16(xa0, *(const v16h*)(wb + (half * 2 + 0) * 512), acc);
      acc = wmma_f16(xa1, *(const v16h*)(wb + (half * 2 + 1) * 512), acc);
      float bb = b1[nt * 16 + nn];
#pragma unroll
      for (int v = 0; v < 8; ++v)
        ldsMid[(v + 8 * g) * 32 + half * 16 + nn] =
            (_Float16)fmaxf(acc[v] + bb, 0.f);
    }
    __syncthreads();
    v16h ma = pack_a(ldsMid, 32, lane);
#pragma unroll
    for (int nt2 = 0; nt2 < 3; ++nt2)
      acc2[nt2] = wmma_f16(ma, *(const v16h*)(wb + (4 + nt2) * 512), acc2[nt2]);
    __syncthreads();
  }

#pragma unroll
  for (int nt2 = 0; nt2 < 3; ++nt2) {
    int col = nt2 * 16 + nn;
    if (col < DMODEL) {
      float bb = b2[col];
#pragma unroll
      for (int v = 0; v < 8; ++v) {
        int row = tm * 16 + v + 8 * g;
        if (row < S) out[(size_t)row * DMODEL + col] = acc2[nt2][v] + bb;
      }
    }
  }
}

// ---------------- residual + LayerNorm; writes f32 + padded f16 --------------

__global__ void addln_kernel(const float* __restrict__ x,
                             const float* __restrict__ rsd,
                             const float* __restrict__ g,
                             const float* __restrict__ b,
                             float* __restrict__ y, _Float16* __restrict__ yh,
                             int S) {
  int t = blockIdx.x * blockDim.x + threadIdx.x;
  if (t >= S_PAD) return;
  if (t >= S) {
#pragma unroll
    for (int d = 0; d < 64; ++d) yh[(size_t)t * 64 + d] = (_Float16)0.f;
    return;
  }
  float buf[DMODEL];
  float mean = 0.f;
#pragma unroll
  for (int d = 0; d < DMODEL; ++d) {
    buf[d] = x[(size_t)t * DMODEL + d] + rsd[(size_t)t * DMODEL + d];
    mean += buf[d];
  }
  mean *= (1.f / 34.f);
  float var = 0.f;
#pragma unroll
  for (int d = 0; d < DMODEL; ++d) {
    float dd = buf[d] - mean;
    var += dd * dd;
  }
  var *= (1.f / 34.f);
  float inv = rsqrtf(var + 1e-5f);
#pragma unroll
  for (int d = 0; d < 64; ++d) {
    float o = 0.f;
    if (d < DMODEL) {
      o = (buf[d] - mean) * inv * g[d] + b[d];
      y[(size_t)t * DMODEL + d] = o;
    }
    yh[(size_t)t * 64 + d] = (_Float16)o;
  }
}

// ---------------- final: mean of agent's 81 token rows -----------------------

__global__ void final_kernel(const float* __restrict__ X,
                             const int* __restrict__ agent,
                             float* __restrict__ out) {
  int d = threadIdx.x;
  if (d >= DMODEL) return;
  size_t start = (size_t)(*agent) * HW;
  float s = 0.f;
  for (int r = 0; r < HW; ++r) s += X[(start + r) * DMODEL + d];
  out[d] = s * (1.f / 81.f);
}

// ---------------- host orchestration ----------------------------------------

extern "C" void kernel_launch(void* const* d_in, const int* in_sizes, int n_in,
                              void* d_out, int out_size, void* d_ws,
                              size_t ws_size, hipStream_t stream) {
  (void)in_sizes; (void)n_in; (void)out_size; (void)ws_size;
  const float* obs = (const float*)d_in[0];
  const float* c1w = (const float*)d_in[1];
  const float* c1b = (const float*)d_in[2];
  const float* c2w = (const float*)d_in[3];
  const float* c2b = (const float*)d_in[4];
  const float* ipw = (const float*)d_in[5];
  const float* ipb = (const float*)d_in[6];
  const float* opw = (const float*)d_in[7];
  const float* opb = (const float*)d_in[8];
  const float* l1w = (const float*)d_in[9];
  const float* l1b = (const float*)d_in[10];
  const float* l2w = (const float*)d_in[11];
  const float* l2b = (const float*)d_in[12];
  const float* g1 = (const float*)d_in[13];
  const float* b1 = (const float*)d_in[14];
  const float* g2 = (const float*)d_in[15];
  const float* b2 = (const float*)d_in[16];
  float* out = (float*)d_out;

  char* ws = (char*)d_ws;
  size_t off = 0;
  auto alloc = [&](size_t bytes) -> char* {
    char* p = ws + off;
    off += (bytes + 255) & ~(size_t)255;
    return p;
  };
  const int S = S_TOK;
  float* cell = (float*)alloc(HW * 16 * sizeof(float));
  int* agent = (int*)alloc(256);
  float* X = (float*)alloc((size_t)S * DMODEL * 4);
  _Float16* Xh = (_Float16*)alloc((size_t)S_PAD * 64 * 2);
  float* qkv = (float*)alloc((size_t)S * 102 * 4);
  _Float16* attOh = (_Float16*)alloc((size_t)S_PAD * 64 * 2);
  float* proj = (float*)alloc((size_t)S * DMODEL * 4);
  float* X1 = (float*)alloc((size_t)S * DMODEL * 4);
  _Float16* X1h = (_Float16*)alloc((size_t)S_PAD * 64 * 2);
  float* ff = (float*)alloc((size_t)S * DMODEL * 4);
  _Float16* ipwS[2], *opwS[2], *W1s[2], *W2s[2];
  for (int l = 0; l < 2; ++l) {
    ipwS[l] = (_Float16*)alloc(7 * 2 * 512 * 2);    // N=102: 7 tiles, K=34: 2 steps
    opwS[l] = (_Float16*)alloc(3 * 2 * 512 * 2);    // N=34: 3 tiles
    W1s[l] = (_Float16*)alloc(128 * 2 * 512 * 2);   // N=2048: 128 tiles
    W2s[l] = (_Float16*)alloc(3 * 64 * 512 * 2);    // N=34, K=2048: 64 steps
  }

  // weight pre-swizzle
  for (int l = 0; l < 2; ++l) {
    int t;
    t = 7 * 2 * 512;
    prep_w_kernel<<<(t + 255) / 256, 256, 0, stream>>>(
        ipw + (size_t)l * 102 * 34, ipwS[l], 102, 34, t);
    t = 3 * 2 * 512;
    prep_w_kernel<<<(t + 255) / 256, 256, 0, stream>>>(
        opw + (size_t)l * 34 * 34, opwS[l], 34, 34, t);
    t = 128 * 2 * 512;
    prep_w_kernel<<<(t + 255) / 256, 256, 0, stream>>>(
        l1w + (size_t)l * 2048 * 34, W1s[l], 2048, 34, t);
    t = 3 * 64 * 512;
    prep_w_kernel<<<(t + 255) / 256, 256, 0, stream>>>(
        l2w + (size_t)l * 34 * 2048, W2s[l], 34, 2048, t);
  }

  conv_kernel<<<1, 256, 0, stream>>>(obs, c1w, c1b, c2w, c2b, cell, agent);
  tokens_kernel<<<(S_PAD * 64 + 255) / 256, 256, 0, stream>>>(cell, X, Xh);
  zeroh_kernel<<<(S_PAD * 64 + 255) / 256, 256, 0, stream>>>(attOh, S_PAD * 64);

  auto tiles = [](int M, int N) { return ((M + 15) / 16) * ((N + 15) / 16); };
  auto gblk = [](int t) { return (t + 7) / 8; };

  const float* Xf = X;
  const _Float16* Xhf = Xh;
  for (int l = 0; l < 2; ++l) {
    gemm_kernel<<<gblk(tiles(S, 102)), 256, 0, stream>>>(
        Xhf, ipwS[l], ipb + l * 102, qkv, S, 102, 2, 0);
    attn_kernel<<<((S + 15) / 16) * 2, 32, 0, stream>>>(qkv, attOh, S);
    gemm_kernel<<<gblk(tiles(S, 34)), 256, 0, stream>>>(
        attOh, opwS[l], opb + l * 34, proj, S, 34, 2, 0);
    addln_kernel<<<(S_PAD + 255) / 256, 256, 0, stream>>>(
        Xf, proj, g1 + l * 34, b1 + l * 34, X1, X1h, S);
    ffn_kernel<<<(S + 15) / 16, 32, 0, stream>>>(X1h, W1s[l], l1b + l * 2048,
                                                 W2s[l], l2b + l * 34, ff, S);
    addln_kernel<<<(S_PAD + 255) / 256, 256, 0, stream>>>(
        X1, ff, g2 + l * 34, b2 + l * 34, X, Xh, S);
    Xf = X;
    Xhf = Xh;
  }
  final_kernel<<<1, 64, 0, stream>>>(X, agent, out);
}
